// MixtureOfAttentionLayer_12893491823136
// MI455X (gfx1250) — matrix operand order repaired and checked by reading, not compiled, not run on hardware
//
#include <hip/hip_runtime.h>
#include <hip/hip_bf16.h>

// ---------------------------------------------------------------------------
// Types for gfx1250 WMMA
// ---------------------------------------------------------------------------
typedef __attribute__((ext_vector_type(16))) __bf16          v16bf;
typedef __attribute__((ext_vector_type(16))) unsigned short  v16us;
typedef __attribute__((ext_vector_type(8)))  float           v8f;
typedef __attribute__((ext_vector_type(4)))  unsigned int    v4u;
typedef int v4i_gcc __attribute__((vector_size(16)));  // matches async builtin

#if __has_builtin(__builtin_amdgcn_global_load_async_to_lds_b128) && \
    __has_builtin(__builtin_amdgcn_s_wait_asynccnt)
#define HAVE_ASYNC_LDS 1
#else
#define HAVE_ASYNC_LDS 0
#endif

__device__ __forceinline__ unsigned int bfround(float f) {
  unsigned int x = __builtin_bit_cast(unsigned int, f);
  return x + 0x7FFFu + ((x >> 16) & 1u);
}
__device__ __forceinline__ unsigned short f2bf(float f) {
  return (unsigned short)(bfround(f) >> 16);
}
__device__ __forceinline__ unsigned int f2bf2(float a, float b) {
#if __has_builtin(__builtin_amdgcn_perm)
  return __builtin_amdgcn_perm(bfround(b), bfround(a), 0x07060302u);
#else
  return (bfround(a) >> 16) | (bfround(b) & 0xFFFF0000u);
#endif
}
__device__ __forceinline__ float bf2f(unsigned short u) {
  return __builtin_bit_cast(float, ((unsigned int)u) << 16);
}

__device__ __forceinline__ v8f wmma_bf16(v16bf a, v16bf b, v8f c) {
  return __builtin_amdgcn_wmma_f32_16x16x32_bf16(
      false, a, false, b, (short)0, c, false, false);
}

union U8x16 { v4u q; unsigned short e[8]; };

__device__ __forceinline__ v16bf frag_from2(v4u lo, v4u hi) {
  union { v4u q[2]; v16us v; } u;
  u.q[0] = lo; u.q[1] = hi;
  return __builtin_bit_cast(v16bf, u.v);
}

__device__ __forceinline__ v16bf a_frag(const unsigned short* t, int ld,
                                        int row0, int koff, int lane) {
  int m  = lane & 15;
  int kb = (lane >> 4) << 3;
  const unsigned short* p = t + (row0 + m) * ld + koff + kb;
  return frag_from2(*(const v4u*)p, *(const v4u*)(p + 16));
}

__device__ __forceinline__ v16bf b_fragT(const unsigned short* t, int ld,
                                         int col0, int koff, int lane) {
  int n  = lane & 15;
  int ko = (lane >> 4) << 3;
  const unsigned short* p = t + (col0 + n) * ld + koff + ko;
  return frag_from2(*(const v4u*)p, *(const v4u*)(p + 16));
}

__device__ __forceinline__ v16bf ones_frag() {
  v16us u;
#pragma unroll
  for (int i = 0; i < 16; ++i) u[i] = 0x3F80;  // bf16 1.0
  return __builtin_bit_cast(v16bf, u);
}

#if HAVE_ASYNC_LDS
__device__ __forceinline__ void async_cp_b128(unsigned short* lds_dst,
                                              const unsigned short* gsrc) {
  __builtin_amdgcn_global_load_async_to_lds_b128(
      (__attribute__((address_space(1))) v4i_gcc*)const_cast<unsigned short*>(gsrc),
      (__attribute__((address_space(3))) v4i_gcc*)lds_dst, 0, 0);
}
#endif

// ---------------------------------------------------------------------------
// Batched GEMM, bf16 operands, f32 accumulate, C(fp32) or Cb(bf16) output.
// 128x64 block tile, K-step 32; wave = 32 rows x 64 cols (8 WMMAs / step).
// FULL: unguarded staging, 32-bit offsets. A operand staged via async
// global->LDS (double buffered) when the builtins exist, else via a VGPR
// prefetch pipeline. B operand: VGPR prefetch + transpose scatter.
// ---------------------------------------------------------------------------
#define ALD 48
#define BMT 128
#define BNT 64
template <bool TA, bool FULL>
__global__ __launch_bounds__(128) void gemm_bf16_kernel(
    const unsigned short* __restrict__ A, long lda, long sA,
    const unsigned short* __restrict__ Bm, long ldb, long sB,
    float* __restrict__ C, unsigned short* __restrict__ Cb,
    long ldc, long sC, const float* __restrict__ bias,
    int M, int N, int K) {
  __shared__ __align__(16) unsigned short At[2][BMT * ALD];
  __shared__ __align__(16) unsigned short Bt[BNT * ALD];
  const int tid = threadIdx.x, w = tid >> 5, lane = tid & 31;
  const int tm = blockIdx.y * BMT, tn = blockIdx.x * BNT;
  A  += (size_t)blockIdx.z * sA;
  Bm += (size_t)blockIdx.z * sB;
  const size_t cbase = (size_t)blockIdx.z * sC;
  const unsigned ldau = (unsigned)lda, ldbu = (unsigned)ldb;

  v8f acc[2][4];
#pragma unroll
  for (int a2 = 0; a2 < 2; ++a2)
#pragma unroll
    for (int f = 0; f < 4; ++f)
#pragma unroll
      for (int r = 0; r < 8; ++r) acc[a2][f][r] = 0.f;

  // loop-invariant staging indices (32-bit offsets, stepped per K-tile)
  int ar[4], ak[4], bk[2], bn8[2];
  unsigned aoff[4], boff[2], astep = 0;
  if constexpr (FULL) {
    if constexpr (!TA) {
#pragma unroll
      for (int c = 0; c < 4; ++c) {
        int chunk = tid + c * 128;             // 0..511
        ar[c] = chunk >> 2; ak[c] = (chunk & 3) * 8;
        aoff[c] = (unsigned)(tm + ar[c]) * ldau + ak[c];
      }
      astep = 32u;
    } else {
#pragma unroll
      for (int c = 0; c < 4; ++c) {
        int chunk = tid + c * 128;
        ak[c] = chunk >> 4; ar[c] = (chunk & 15) * 8;  // ar = r8
        aoff[c] = (unsigned)ak[c] * ldau + (unsigned)(tm + ar[c]);
      }
      astep = 32u * ldau;
    }
#pragma unroll
    for (int c = 0; c < 2; ++c) {
      int chunk = tid + c * 128;               // 0..255
      bk[c] = chunk >> 3; bn8[c] = (chunk & 7) * 8;
      boff[c] = (unsigned)bk[c] * ldbu + (unsigned)(tn + bn8[c]);
    }
  }

  v4u arg[4], brg[2];
  auto fetchB = [&]() {
#pragma unroll
    for (int c = 0; c < 2; ++c) {
      brg[c] = *(const v4u*)(Bm + boff[c]);
      boff[c] += 32u * ldbu;
      __builtin_prefetch((const void*)(Bm + boff[c]), 0, 1);  // 2 tiles ahead
    }
  };
  auto storeB = [&]() {
#pragma unroll
    for (int c = 0; c < 2; ++c) {
      U8x16 g; g.q = brg[c];
#pragma unroll
      for (int j = 0; j < 8; ++j) Bt[(bn8[c] + j) * ALD + bk[c]] = g.e[j];
    }
  };
  auto compute_tile = [&](const unsigned short* Atile) {
    v16bf af[2];
    af[0] = a_frag(Atile, ALD, w * 32, 0, lane);
    af[1] = a_frag(Atile, ALD, w * 32 + 16, 0, lane);
    v16bf bfr[4];
#pragma unroll
    for (int f = 0; f < 4; ++f) bfr[f] = b_fragT(Bt, ALD, f * 16, 0, lane);
#pragma unroll
    for (int a2 = 0; a2 < 2; ++a2)
#pragma unroll
      for (int f = 0; f < 4; ++f)
        acc[a2][f] = wmma_bf16(af[a2], bfr[f], acc[a2][f]);
  };

  if constexpr (FULL) {
    const int nt = K / 32;
#if HAVE_ASYNC_LDS
    if constexpr (!TA) {
      // A operand: async global->LDS, double buffered.
      auto issueA = [&](int buf) {
#pragma unroll
        for (int c = 0; c < 4; ++c) {
          async_cp_b128(&At[buf][ar[c] * ALD + ak[c]], A + aoff[c]);
          aoff[c] += 32u;
        }
      };
      issueA(0);
      fetchB();
      for (int kt = 0; kt < nt; ++kt) {
        const int cur = kt & 1;
        __syncthreads();  // all waves done reading At[cur^1] and Bt
        if (kt + 1 < nt) issueA(cur ^ 1);  // overlaps with compute below
        storeB();
        if (kt + 1 < nt) __builtin_amdgcn_s_wait_asynccnt(4);  // tile kt done
        else             __builtin_amdgcn_s_wait_asynccnt(0);
        __syncthreads();
        if (kt + 1 < nt) fetchB();
        compute_tile(At[cur]);
      }
    } else
#endif
    {
      // VGPR prefetch pipeline
      auto fetchA = [&]() {
#pragma unroll
        for (int c = 0; c < 4; ++c) {
          arg[c] = *(const v4u*)(A + aoff[c]);
          aoff[c] += astep;
        }
      };
      fetchA();
      fetchB();
      for (int kt = 0; kt < nt; ++kt) {
        __syncthreads();
        if constexpr (!TA) {
#pragma unroll
          for (int c = 0; c < 4; ++c)
            *(v4u*)(&At[0][ar[c] * ALD + ak[c]]) = arg[c];
        } else {
#pragma unroll
          for (int c = 0; c < 4; ++c) {
            U8x16 g; g.q = arg[c];
#pragma unroll
            for (int j = 0; j < 8; ++j) At[0][(ar[c] + j) * ALD + ak[c]] = g.e[j];
          }
        }
        storeB();
        __syncthreads();
        if (kt + 1 < nt) { fetchA(); fetchB(); }
        compute_tile(At[0]);
      }
    }
  } else {
    // branchless guarded staging (clamped addresses + selects)
    for (int k0 = 0; k0 < K; k0 += 32) {
      __syncthreads();
#pragma unroll
      for (int c = 0; c < 4; ++c) {
        int chunk = tid + c * 128;
        int r = chunk >> 2, kk3 = (chunk & 3) * 8;
        int gr = tm + r;
        int cgr = gr < M ? gr : (M - 1);
        unsigned short e[8];
#pragma unroll
        for (int u2 = 0; u2 < 8; ++u2) {
          int gk = k0 + kk3 + u2;
          int cgk = gk < K ? gk : (K - 1);
          unsigned short vv = TA ? A[(unsigned)cgk * ldau + (unsigned)cgr]
                                 : A[(unsigned)cgr * ldau + (unsigned)cgk];
          e[u2] = (gr < M && gk < K) ? vv : (unsigned short)0;
        }
        v4u pk;
        pk.x = e[0] | ((unsigned)e[1] << 16);
        pk.y = e[2] | ((unsigned)e[3] << 16);
        pk.z = e[4] | ((unsigned)e[5] << 16);
        pk.w = e[6] | ((unsigned)e[7] << 16);
        *(v4u*)(&At[0][r * ALD + kk3]) = pk;
      }
#pragma unroll
      for (int c = 0; c < 8; ++c) {
        int p = tid + c * 128;
        int n = p & 63, k2 = p >> 6;
        int gk = k0 + 2 * k2, gc = tn + n;
        int cg0 = gk < K ? gk : (K - 1);
        int cg1 = (gk + 1) < K ? (gk + 1) : (K - 1);
        int cgc = gc < N ? gc : (N - 1);
        unsigned short b0 = Bm[(unsigned)cg0 * ldbu + (unsigned)cgc];
        unsigned short b1 = Bm[(unsigned)cg1 * ldbu + (unsigned)cgc];
        b0 = (gk < K && gc < N) ? b0 : (unsigned short)0;
        b1 = ((gk + 1) < K && gc < N) ? b1 : (unsigned short)0;
        *(unsigned int*)(&Bt[n * ALD + 2 * k2]) =
            (unsigned)b0 | ((unsigned)b1 << 16);
      }
      __syncthreads();
      compute_tile(At[0]);
    }
  }

  int n = lane & 15, mo = (lane >> 4) << 3;
#pragma unroll
  for (int a2 = 0; a2 < 2; ++a2)
#pragma unroll
    for (int f = 0; f < 4; ++f)
#pragma unroll
      for (int r = 0; r < 8; ++r) {
        int row = tm + w * 32 + a2 * 16 + r + mo;
        int col = tn + f * 16 + n;
        if (FULL || (row < M && col < N)) {
          float v = acc[a2][f][r];
          if (bias) v += bias[col];
          if (Cb) Cb[cbase + (size_t)row * ldc + col] = f2bf(v);
          else    C [cbase + (size_t)row * ldc + col] = v;
        }
      }
}

// ---------------------------------------------------------------------------
// Flash attention. Q,K,V bf16 padded [B*nh][S][128]; Q pre-scaled.
// Softmax row sums via WMMA against a constant ones-B fragment.
// ---------------------------------------------------------------------------
#define KLD 136
#define VLD 40
#define QLD 136
template <int LOCAL>
__global__ __launch_bounds__(128) void flash_kernel(
    const unsigned short* __restrict__ Qp, const unsigned short* __restrict__ Kp,
    const unsigned short* __restrict__ Vp, const int* __restrict__ amask,
    float* __restrict__ Op, int S, int nh) {
  __shared__ __align__(16) unsigned short Kt[32 * KLD];
  __shared__ __align__(16) unsigned short Vt[128 * VLD];
  __shared__ __align__(16) unsigned short Qt[4][16 * QLD];
  __shared__ __align__(16) unsigned short Pt[4][16 * 32];

  const int w = threadIdx.x >> 5, lane = threadIdx.x & 31;
  const int bh = blockIdx.y, b = bh / nh;
  const size_t base = (size_t)bh * S * 128;
  const unsigned short* Qb = Qp + base;
  const unsigned short* Kb = Kp + base;
  const unsigned short* Vb = Vp + base;
  float* Ob = Op + base;
  const int q0 = blockIdx.x * 64 + w * 16;

#pragma unroll
  for (int c = 0; c < 8; ++c) {
    int chunk = lane + c * 32;
    int m = chunk >> 4, d8 = (chunk & 15) * 8;
    *(v4u*)(Qt[w] + m * QLD + d8) =
        *(const v4u*)(Qb + (unsigned)(q0 + m) * 128u + d8);
  }
  v16bf qf[4];
#pragma unroll
  for (int c = 0; c < 4; ++c) qf[c] = a_frag(Qt[w], QLD, 0, 32 * c, lane);
  const v16bf ones = ones_frag();

  v8f acc[8];
  float mrow[8], lrow[8];
#pragma unroll
  for (int r = 0; r < 8; ++r) { mrow[r] = -1e30f; lrow[r] = 0.f; }
#pragma unroll
  for (int f = 0; f < 8; ++f)
#pragma unroll
    for (int r = 0; r < 8; ++r) acc[f][r] = 0.f;

  int skey[4], sd8[4];
#pragma unroll
  for (int c = 0; c < 4; ++c) {
    int chunk = threadIdx.x + c * 128;
    skey[c] = chunk >> 4; sd8[c] = (chunk & 15) * 8;
  }

  const int ntile = S / 32;
  for (int kt = 0; kt < ntile; ++kt) {
    const int k0 = kt * 32;
    __syncthreads();
#pragma unroll
    for (int c = 0; c < 4; ++c) {
      unsigned gi = (unsigned)(k0 + skey[c]) * 128u + sd8[c];
      *(v4u*)(Kt + skey[c] * KLD + sd8[c]) = *(const v4u*)(Kb + gi);
      U8x16 g;
      g.q = *(const v4u*)(Vb + gi);
#pragma unroll
      for (int j = 0; j < 8; ++j) Vt[(sd8[c] + j) * VLD + skey[c]] = g.e[j];
    }
    __syncthreads();

    bool active = true;
    if (LOCAL) active = (k0 <= q0 + 15 + 32) && (k0 + 31 >= q0 - 32);
    if (active) {
      v8f s0, s1;
#pragma unroll
      for (int r = 0; r < 8; ++r) { s0[r] = 0.f; s1[r] = 0.f; }
#pragma unroll
      for (int c = 0; c < 4; ++c) {
        v16bf b0 = b_fragT(Kt, KLD, 0, 32 * c, lane);
        v16bf b1 = b_fragT(Kt, KLD, 16, 32 * c, lane);
        s0 = wmma_bf16(qf[c], b0, s0);
        s1 = wmma_bf16(qf[c], b1, s1);
      }
      const int col0 = k0 + (lane & 15);
      const int am0 = amask[b * S + col0];
      const int am1 = amask[b * S + col0 + 16];
      const int rowbase = q0 + ((lane >> 4) << 3);
      float alf[8];
#pragma unroll
      for (int r = 0; r < 8; ++r) {
        float v0 = s0[r], v1 = s1[r];
        if (am0 == 0) v0 = -1e30f;
        if (am1 == 0) v1 = -1e30f;
        if (LOCAL) {
          int qr = rowbase + r;
          int d0 = col0 - qr;      if (d0 < 0) d0 = -d0;
          int d1 = col0 + 16 - qr; if (d1 < 0) d1 = -d1;
          if (d0 > 32) v0 = -1e30f;
          if (d1 > 32) v1 = -1e30f;
        }
        float mx = fmaxf(v0, v1);
#pragma unroll
        for (int d = 1; d < 16; d <<= 1) mx = fmaxf(mx, __shfl_xor(mx, d, 32));
        float mn = fmaxf(mrow[r], mx);
        alf[r] = __expf(mrow[r] - mn);
        mrow[r] = mn;
        float p0 = __expf(v0 - mn), p1 = __expf(v1 - mn);
#pragma unroll
        for (int f = 0; f < 8; ++f) acc[f][r] *= alf[r];
        int pr = (r + ((lane >> 4) << 3)) * 32;
        Pt[w][pr + (lane & 15)]      = f2bf(p0);
        Pt[w][pr + 16 + (lane & 15)] = f2bf(p1);
      }
      v16bf pf = a_frag(Pt[w], 32, 0, 0, lane);
      v8f zs;
#pragma unroll
      for (int r = 0; r < 8; ++r) zs[r] = 0.f;
      v8f sfr = wmma_bf16(pf, ones, zs);
#pragma unroll
      for (int r = 0; r < 8; ++r) lrow[r] = lrow[r] * alf[r] + sfr[r];
      v16bf vfr[8];
#pragma unroll
      for (int f = 0; f < 8; ++f) vfr[f] = b_fragT(Vt, VLD, f * 16, 0, lane);
#pragma unroll
      for (int f = 0; f < 8; ++f) acc[f] = wmma_bf16(pf, vfr[f], acc[f]);
    }
  }

  const int n = lane & 15, mo = (lane >> 4) << 3;
#pragma unroll
  for (int r = 0; r < 8; ++r) {
    float inv = 1.0f / (lrow[r] + 1e-30f);
#pragma unroll
    for (int f = 0; f < 8; ++f)
      Ob[(unsigned)(q0 + r + mo) * 128u + f * 16 + n] = acc[f][r] * inv;
  }
}

// ---------------------------------------------------------------------------
// Small helper kernels
// ---------------------------------------------------------------------------
__global__ void cvt_bf16_kernel(const float* __restrict__ src,
                                unsigned short* __restrict__ dst, long n2) {
  long i = blockIdx.x * (long)blockDim.x + threadIdx.x;
  if (i >= n2) return;
  ((unsigned int*)dst)[i] = f2bf2(src[2 * i], src[2 * i + 1]);
}

__global__ void pack_heads_kernel(const float* __restrict__ src,
                                  unsigned short* __restrict__ dst,
                                  int nh, int hd, int S, int total,
                                  float scale) {
  int i = blockIdx.x * blockDim.x + threadIdx.x;
  if (i >= total) return;
  int d = i & 127;
  int t = i >> 7;
  int s = t % S;
  int t2 = t / S;
  int h = t2 % nh;
  int b = t2 / nh;
  float v = 0.f;
  if (d < hd) v = src[((size_t)(b * S + s)) * (nh * hd) + h * hd + d] * scale;
  dst[i] = f2bf(v);
}

__global__ void unpack_heads_kernel(const float* __restrict__ src,
                                    unsigned short* __restrict__ dst,
                                    int nh, int hd, int S, int total) {
  int i = blockIdx.x * blockDim.x + threadIdx.x;
  if (i >= total) return;
  int c = i & 1023;
  int row = i >> 10;
  int pd = nh * hd;
  float v = 0.f;
  if (c < pd) {
    int h = c / hd, d = c % hd;
    int b = row / S, s = row % S;
    v = src[(((size_t)(b * nh + h)) * S + s) * 128 + d];
  }
  dst[i] = f2bf(v);
}

__global__ void elu1_bf16_kernel(unsigned short* __restrict__ p, int n) {
  int i = blockIdx.x * blockDim.x + threadIdx.x;
  if (i < n) {
    float x = bf2f(p[i]);
    p[i] = f2bf((x > 0.f) ? (x + 1.f) : __expf(x));
  }
}

__global__ void ksum_kernel(const unsigned short* __restrict__ K,
                            float* __restrict__ ks, int S) {
  int bh = blockIdx.x, d = threadIdx.x;
  const unsigned short* kb = K + (size_t)bh * S * 128;
  float s = 0.f;
  for (int t = 0; t < S; ++t) s += bf2f(kb[(unsigned)t * 128u + d]);
  ks[bh * 128 + d] = s;
}

__global__ void linrow_kernel(const unsigned short* __restrict__ Q,
                              const float* __restrict__ ks,
                              float* __restrict__ num, int S, int total_rows) {
  int w = (blockIdx.x * blockDim.x + threadIdx.x) >> 5;
  int lane = threadIdx.x & 31;
  if (w >= total_rows) return;
  int bh = w / S;
  const unsigned short* q = Q + (size_t)w * 128;
  const float* k = ks + bh * 128;
  float d = 0.f;
  for (int i = lane; i < 128; i += 32) d += bf2f(q[i]) * k[i];
#pragma unroll
  for (int m = 1; m < 32; m <<= 1) d += __shfl_xor(d, m, 32);
  d += 1e-6f;
  float inv = 1.0f / d;
  float* o = num + (size_t)w * 128;
  for (int i = lane; i < 128; i += 32) o[i] *= inv;
}

__global__ void gate_kernel(const float* __restrict__ x,
                            const float* __restrict__ Wg,
                            float* __restrict__ wrow, int M, int H) {
  int row = blockIdx.x * blockDim.x + threadIdx.x;
  if (row >= M) return;
  float lg[4] = {0.f, 0.f, 0.f, 0.f};
  const float* xr = x + (size_t)row * H;
  for (int k = 0; k < H; ++k) {
    float xv = xr[k];
#pragma unroll
    for (int e = 0; e < 4; ++e) lg[e] += xv * Wg[k * 4 + e];
  }
  int i0 = 0;
#pragma unroll
  for (int e = 1; e < 4; ++e) if (lg[e] > lg[i0]) i0 = e;
  int i1 = -1;
#pragma unroll
  for (int e = 0; e < 4; ++e)
    if (e != i0 && (i1 < 0 || lg[e] > lg[i1])) i1 = e;
  float a = lg[i0], bb = lg[i1];
  float eb = __expf(bb - a);
  float s = 1.f + eb;
#pragma unroll
  for (int e = 0; e < 4; ++e) wrow[e * M + row] = 0.f;
  wrow[i0 * M + row] = 1.f / s;
  wrow[i1 * M + row] = eb / s;
}

__global__ void accum_kernel(const float* __restrict__ src,
                             const float* __restrict__ wr,
                             float* __restrict__ out, int M, int H) {
  int i = blockIdx.x * blockDim.x + threadIdx.x;
  if (i >= M * H) return;
  int row = i / H;
  out[i] += wr[row] * src[i];
}

// ---------------------------------------------------------------------------
// Host driver
// ---------------------------------------------------------------------------
static void launch_gemm(hipStream_t st, const unsigned short* A, long lda,
                        long sA, bool tA, const unsigned short* B, long ldb,
                        long sB, float* C, unsigned short* Cb, long ldc,
                        long sC, const float* bias, int M, int N, int K,
                        int batch) {
  dim3 g((N + BNT - 1) / BNT, (M + BMT - 1) / BMT, batch);
  bool full = (M % BMT == 0) && (N % BNT == 0) && (K % 32 == 0) &&
              ((lda & 7) == 0) && ((ldb & 7) == 0);
  if (tA) {
    if (full)
      gemm_bf16_kernel<true, true><<<g, 128, 0, st>>>(
          A, lda, sA, B, ldb, sB, C, Cb, ldc, sC, bias, M, N, K);
    else
      gemm_bf16_kernel<true, false><<<g, 128, 0, st>>>(
          A, lda, sA, B, ldb, sB, C, Cb, ldc, sC, bias, M, N, K);
  } else {
    if (full)
      gemm_bf16_kernel<false, true><<<g, 128, 0, st>>>(
          A, lda, sA, B, ldb, sB, C, Cb, ldc, sC, bias, M, N, K);
    else
      gemm_bf16_kernel<false, false><<<g, 128, 0, st>>>(
          A, lda, sA, B, ldb, sB, C, Cb, ldc, sC, bias, M, N, K);
  }
}
static void launch_cvt(hipStream_t st, const float* src, unsigned short* dst,
                       long n) {
  long n2 = n / 2;
  cvt_bf16_kernel<<<(n2 + 255) / 256, 256, 0, st>>>(src, dst, n2);
}

extern "C" void kernel_launch(void* const* d_in, const int* in_sizes, int n_in,
                              void* d_out, int out_size, void* d_ws,
                              size_t ws_size, hipStream_t stream) {
  (void)in_sizes; (void)n_in; (void)ws_size;
  const int B = 4, S = 1024, H = 1024, E = 4;
  const int M = B * S;
  const int cfg_nh[E]   = {8, 12, 8, 8};
  const int cfg_type[E] = {0, 0, 2, 1};  // 0=standard, 1=local, 2=linear

  const float* x  = (const float*)d_in[0];
  const float* Wg = (const float*)d_in[1];
  const int* amask = (const int*)d_in[2 + 8 * E];

  char* base = (char*)d_ws;
  auto alloc = [&](size_t bytes) -> char* {
    char* p = base;
    base += (bytes + 255) & ~(size_t)255;
    return p;
  };
  const size_t NPH = (size_t)B * 12 * S * 128;
  unsigned short* xb   = (unsigned short*)alloc((size_t)M * H * 2);
  unsigned short* wb   = (unsigned short*)alloc((size_t)H * H * 2);
  unsigned short* Qp   = (unsigned short*)alloc(NPH * 2);
  unsigned short* Kp   = (unsigned short*)alloc(NPH * 2);
  unsigned short* Vp   = (unsigned short*)alloc(NPH * 2);
  float*          Opd  = (float*)alloc(NPH * 4);
  float*          tmp  = (float*)alloc((size_t)M * H * 4);
  unsigned short* tmpA = (unsigned short*)alloc((size_t)M * H * 2);
  unsigned short* kvb  = (unsigned short*)alloc((size_t)32 * 128 * 128 * 2);
  float*          ksb  = (float*)alloc((size_t)32 * 128 * 4);
  float*          wrow = (float*)alloc((size_t)E * M * 4);

  float* out = (float*)d_out;
  (void)hipMemsetAsync(out, 0, (size_t)out_size * sizeof(float), stream);

  launch_cvt(stream, x, xb, (long)M * H);
  gate_kernel<<<(M + 127) / 128, 128, 0, stream>>>(x, Wg, wrow, M, H);

  for (int e = 0; e < E; ++e) {
    const int nh = cfg_nh[e];
    const int at = cfg_type[e];
    const int hd = H / nh;
    const int pd = nh * hd;
    const float* Wq = (const float*)d_in[2 + 8 * e + 0];
    const float* bq = (const float*)d_in[2 + 8 * e + 1];
    const float* Wk = (const float*)d_in[2 + 8 * e + 2];
    const float* bk = (const float*)d_in[2 + 8 * e + 3];
    const float* Wv = (const float*)d_in[2 + 8 * e + 4];
    const float* bv = (const float*)d_in[2 + 8 * e + 5];
    const float* Wo = (const float*)d_in[2 + 8 * e + 6];
    const float* bo = (const float*)d_in[2 + 8 * e + 7];

    const int tot_pad = B * nh * S * 128;
    const float qscale = (at == 2) ? 1.f : rsqrtf((float)hd);

    launch_cvt(stream, Wq, wb, (long)H * pd);
    launch_gemm(stream, xb, H, 0, false, wb, pd, 0, tmp, nullptr, pd, 0, bq,
                M, pd, H, 1);
    pack_heads_kernel<<<(tot_pad + 255) / 256, 256, 0, stream>>>(
        tmp, Qp, nh, hd, S, tot_pad, qscale);
    launch_cvt(stream, Wk, wb, (long)H * pd);
    launch_gemm(stream, xb, H, 0, false, wb, pd, 0, tmp, nullptr, pd, 0, bk,
                M, pd, H, 1);
    pack_heads_kernel<<<(tot_pad + 255) / 256, 256, 0, stream>>>(
        tmp, Kp, nh, hd, S, tot_pad, 1.f);
    launch_cvt(stream, Wv, wb, (long)H * pd);
    launch_gemm(stream, xb, H, 0, false, wb, pd, 0, tmp, nullptr, pd, 0, bv,
                M, pd, H, 1);
    pack_heads_kernel<<<(tot_pad + 255) / 256, 256, 0, stream>>>(
        tmp, Vp, nh, hd, S, tot_pad, 1.f);

    if (at == 2) {
      const int BH = B * nh;
      elu1_bf16_kernel<<<(tot_pad + 255) / 256, 256, 0, stream>>>(Qp, tot_pad);
      elu1_bf16_kernel<<<(tot_pad + 255) / 256, 256, 0, stream>>>(Kp, tot_pad);
      ksum_kernel<<<BH, 128, 0, stream>>>(Kp, ksb, S);
      launch_gemm(stream, Kp, 128, (long)S * 128, true, Vp, 128, (long)S * 128,
                  nullptr, kvb, 128, 128 * 128, nullptr, 128, 128, S, BH);
      launch_gemm(stream, Qp, 128, (long)S * 128, false, kvb, 128, 128 * 128,
                  Opd, nullptr, 128, (long)S * 128, nullptr, S, 128, 128, BH);
      int rows = BH * S;
      linrow_kernel<<<(rows * 32 + 127) / 128, 128, 0, stream>>>(
          Qp, ksb, Opd, S, rows);
    } else {
      dim3 g(S / 64, B * nh);
      if (at == 1)
        flash_kernel<1><<<g, 128, 0, stream>>>(Qp, Kp, Vp, amask, Opd, S, nh);
      else
        flash_kernel<0><<<g, 128, 0, stream>>>(Qp, Kp, Vp, amask, Opd, S, nh);
    }

    const int tot_un = M * 1024;
    unpack_heads_kernel<<<(tot_un + 255) / 256, 256, 0, stream>>>(
        Opd, tmpA, nh, hd, S, tot_un);
    launch_cvt(stream, Wo, wb, (long)pd * H);
    launch_gemm(stream, tmpA, 1024, 0, false, wb, H, 0, tmp, nullptr, H, 0, bo,
                M, H, pd, 1);
    accum_kernel<<<((size_t)M * H + 255) / 256, 256, 0, stream>>>(
        tmp, wrow + (size_t)e * M, out, M, H);
  }
}
